// Attention_79748952752529
// MI455X (gfx1250) — compile-verified, matrix operand order checked by reading, not run on hardware
//
#include <hip/hip_runtime.h>
#include <hip/hip_bf16.h>

typedef __bf16 bf16_t;
typedef __attribute__((ext_vector_type(16))) __bf16 v16bf;
typedef __attribute__((ext_vector_type(8)))  __bf16 v8bf;
typedef __attribute__((ext_vector_type(8)))  float  v8f;

#define B_  8
#define T_  1024
#define DIM_ 1024
#define H_  16
#define D_  64
#define NEGV -1.0e9f

__device__ __forceinline__ bf16_t f2bf(float f) {
  union { float f; unsigned u; } x; x.f = f;
  unsigned r = (x.u + 0x7FFFu + ((x.u >> 16) & 1u)) >> 16;
  union { unsigned short s; bf16_t b; } y; y.s = (unsigned short)r;
  return y.b;
}

// Build a v16bf A/B fragment from two contiguous 8-element (16B) chunks.
__device__ __forceinline__ v16bf load_frag(const bf16_t* p0, const bf16_t* p1) {
  v8bf lo = *(const v8bf*)p0;
  v8bf hi = *(const v8bf*)p1;
  return __builtin_shufflevector(lo, hi, 0,1,2,3,4,5,6,7,8,9,10,11,12,13,14,15);
}

__device__ __forceinline__ v8f wmma_bf16(v16bf a, v16bf b, v8f c) {
  return __builtin_amdgcn_wmma_f32_16x16x32_bf16(false, a, false, b, (short)0, c, false, false);
}

// ---------------------------------------------------------------------------
// Projection: Out = X @ W^T + bias, X:[B*T,DIM] f32, W:[DIM,DIM] f32 (row n is
// contiguous in k -> NT GEMM). Output bf16 in [B,H,T,64] (transpose_out=0) or
// [B,H,64,T] (transpose_out=1, used for V).
// Block: 128x128 tile, 8 waves (256 thr), wave computes 32(M)x64(N).
// ---------------------------------------------------------------------------
__global__ __launch_bounds__(256)
void proj_kernel(const float* __restrict__ X, const float* __restrict__ W,
                 const float* __restrict__ bias, bf16_t* __restrict__ Out,
                 int transpose_out)
{
  __shared__ bf16_t sA[128 * 32];
  __shared__ bf16_t sW[128 * 32];
  const int tid  = threadIdx.x;
  const int wave = tid >> 5;
  const int lane = tid & 31;
  const int half = lane >> 4;
  const int ln16 = lane & 15;
  const int wm = wave >> 1;          // 0..3  -> 32-row sub-tile
  const int wn = wave & 1;           // 0..1  -> 64-col sub-tile
  const int m0 = blockIdx.x * 128;
  const int n0 = blockIdx.y * 128;

  v8f acc[2][4];
#pragma unroll
  for (int i = 0; i < 2; i++)
#pragma unroll
    for (int j = 0; j < 4; j++)
#pragma unroll
      for (int r = 0; r < 8; r++) acc[i][j][r] = 0.0f;

  for (int kb = 0; kb < DIM_; kb += 32) {
    // cooperative f32 -> bf16 tile staging (128 rows x 32 cols, 8 float4/row)
    for (int s = tid; s < 128 * 8; s += 256) {
      const int row = s >> 3, c4 = (s & 7) * 4;
      const float4 a4 = *(const float4*)(X + (size_t)(m0 + row) * DIM_ + kb + c4);
      bf16_t* da = &sA[row * 32 + c4];
      da[0] = f2bf(a4.x); da[1] = f2bf(a4.y); da[2] = f2bf(a4.z); da[3] = f2bf(a4.w);
      const float4 w4 = *(const float4*)(W + (size_t)(n0 + row) * DIM_ + kb + c4);
      bf16_t* dw = &sW[row * 32 + c4];
      dw[0] = f2bf(w4.x); dw[1] = f2bf(w4.y); dw[2] = f2bf(w4.z); dw[3] = f2bf(w4.w);
    }
    __syncthreads();

    v16bf a[2], b[4];
#pragma unroll
    for (int i = 0; i < 2; i++) {
      const bf16_t* base = &sA[(wm * 32 + i * 16 + ln16) * 32];
      a[i] = load_frag(base + half * 8, base + 16 + half * 8);
    }
#pragma unroll
    for (int j = 0; j < 4; j++) {
      const bf16_t* base = &sW[(wn * 64 + j * 16 + ln16) * 32];
      b[j] = load_frag(base + half * 8, base + 16 + half * 8);
    }
#pragma unroll
    for (int i = 0; i < 2; i++)
#pragma unroll
      for (int j = 0; j < 4; j++) acc[i][j] = wmma_bf16(a[i], b[j], acc[i][j]);
    __syncthreads();
  }

  // epilogue: bias add + head-split store
#pragma unroll
  for (int j = 0; j < 4; j++) {
    const int n = n0 + wn * 64 + j * 16 + ln16;
    const float bn = bias[n];
    const int h = n >> 6, d = n & 63;
#pragma unroll
    for (int i = 0; i < 2; i++) {
#pragma unroll
      for (int r = 0; r < 8; r++) {
        const int m = m0 + wm * 32 + i * 16 + r + half * 8;
        const int bb = m >> 10, t = m & 1023;
        const float v = acc[i][j][r] + bn;
        const size_t idx = transpose_out
            ? ((size_t)(bb * H_ + h) * D_ + d) * T_ + t
            : ((size_t)(bb * H_ + h) * T_ + t) * D_ + d;
        Out[idx] = f2bf(v);
      }
    }
  }
}

// ---------------------------------------------------------------------------
// Fused attention: one workgroup = (b, h, 128 q rows), 8 waves, wave owns 16
// q rows. Streams 64-key blocks: K/V tiles staged once per workgroup in LDS
// (shared by 8 waves), online softmax, out = softmax(QK^T*s, mask)@V + beta@V.
// Q/K: [B,H,T,64] bf16, V: [B,H,64,T] bf16 (transposed).
// ---------------------------------------------------------------------------
__global__ __launch_bounds__(256)
void attn_kernel(const bf16_t* __restrict__ Qh, const bf16_t* __restrict__ Kh,
                 const bf16_t* __restrict__ Vt, const float* __restrict__ beta,
                 const unsigned char* __restrict__ srcm,
                 const unsigned char* __restrict__ tgtm,
                 float* __restrict__ Out)
{
  __shared__ bf16_t sK[64 * 64];      // K rows tk, cols d      (8 KB)
  __shared__ bf16_t sV[64 * 64];      // V^T rows d, cols tk    (8 KB)
  __shared__ bf16_t sP[8][16 * 64];   // per-wave P patches     (16 KB)
  const int tid  = threadIdx.x;
  const int lane = tid & 31;
  const int wave = tid >> 5;          // 0..7
  const int half = lane >> 4;
  const int ln16 = lane & 15;
  const int q0 = blockIdx.x * 128 + wave * 16;
  const int h  = blockIdx.y;
  const int b  = blockIdx.z;
  const float scale = 0.03125f; // DIM^-0.5 = 1/32

  const size_t bh = (size_t)(b * H_ + h);
  const bf16_t* Q = Qh + bh * (T_ * D_);
  const bf16_t* K = Kh + bh * (T_ * D_);
  const bf16_t* V = Vt + bh * (D_ * T_);
  const float*  Bt = beta + (size_t)h * T_ * T_;

  // persistent Q fragments (rows q0+ln16, d-chunks 0-31 / 32-63)
  v16bf aq[2];
  {
    const bf16_t* qr = Q + (size_t)(q0 + ln16) * D_;
    aq[0] = load_frag(qr + half * 8,      qr + 16 + half * 8);
    aq[1] = load_frag(qr + 32 + half * 8, qr + 48 + half * 8);
  }
  bool tv[8];
#pragma unroll
  for (int j = 0; j < 8; j++) tv[j] = tgtm[b * T_ + q0 + j + half * 8] != 0;

  v8f O[4], Bc[4];
#pragma unroll
  for (int t = 0; t < 4; t++)
#pragma unroll
    for (int j = 0; j < 8; j++) { O[t][j] = 0.0f; Bc[t][j] = 0.0f; }
  float mrun[8], lrun[8];
#pragma unroll
  for (int j = 0; j < 8; j++) { mrun[j] = -3.0e38f; lrun[j] = 0.0f; }

  bf16_t* myP = sP[wave];

  for (int tk0 = 0; tk0 < T_; tk0 += 64) {
    // ---- cooperative K/V tile staging (shared by all 8 waves) ----------
    // 64 rows x 64 bf16 = 512 uint4 chunks each; 2 per thread.
#pragma unroll
    for (int it = 0; it < 2; it++) {
      const int idx = tid + it * 256;         // 0..511
      const int row = idx >> 3, c8 = (idx & 7) * 8;
      *(uint4*)&sK[row * 64 + c8] = *(const uint4*)(K + (size_t)(tk0 + row) * D_ + c8);
      *(uint4*)&sV[row * 64 + c8] = *(const uint4*)(V + (size_t)row * T_ + tk0 + c8);
    }
    if (tk0 + 64 < T_) {  // prefetch next tiles toward WGP$ (global_prefetch_b8)
      __builtin_prefetch(K + (size_t)(tk0 + 64 + (tid >> 2)) * D_ + (tid & 3) * 16, 0, 1);
      __builtin_prefetch(V + (size_t)(tid >> 2) * T_ + tk0 + 64 + (tid & 3) * 16, 0, 1);
    }
    __syncthreads();
    // ---- S = Q K^T for 16x64 block -------------------------------------
    v8f s[4];
#pragma unroll
    for (int t = 0; t < 4; t++)
#pragma unroll
      for (int j = 0; j < 8; j++) s[t][j] = 0.0f;
#pragma unroll
    for (int dk = 0; dk < 2; dk++) {
#pragma unroll
      for (int t = 0; t < 4; t++) {
        const bf16_t* kr = &sK[(t * 16 + ln16) * 64 + dk * 32];
        v16bf bk = load_frag(kr + half * 8, kr + 16 + half * 8);
        s[t] = wmma_bf16(aq[dk], bk, s[t]);
      }
    }
    // ---- scale + combined mask -----------------------------------------
    bool sm[4];
#pragma unroll
    for (int t = 0; t < 4; t++) sm[t] = srcm[b * T_ + tk0 + t * 16 + ln16] != 0;
#pragma unroll
    for (int t = 0; t < 4; t++)
#pragma unroll
      for (int j = 0; j < 8; j++)
        s[t][j] = (sm[t] && tv[j]) ? s[t][j] * scale : NEGV;
    // ---- online softmax (row = fixed VGPR j + lane-half; 16-lane reduce)
    float f[8];
#pragma unroll
    for (int j = 0; j < 8; j++) {
      float mx = fmaxf(fmaxf(s[0][j], s[1][j]), fmaxf(s[2][j], s[3][j]));
#pragma unroll
      for (int off = 8; off; off >>= 1) mx = fmaxf(mx, __shfl_xor(mx, off, 32));
      const float mnew = fmaxf(mrun[j], mx);
      f[j] = __expf(mrun[j] - mnew);
      mrun[j] = mnew;
    }
    float rs[8];
#pragma unroll
    for (int j = 0; j < 8; j++) rs[j] = 0.0f;
#pragma unroll
    for (int t = 0; t < 4; t++)
#pragma unroll
      for (int j = 0; j < 8; j++) {
        const float p = __expf(s[t][j] - mrun[j]);
        s[t][j] = p;
        rs[j] += p;
      }
#pragma unroll
    for (int j = 0; j < 8; j++) {
#pragma unroll
      for (int off = 8; off; off >>= 1) rs[j] += __shfl_xor(rs[j], off, 32);
      lrun[j] = lrun[j] * f[j] + rs[j];
    }
#pragma unroll
    for (int t = 0; t < 4; t++)
#pragma unroll
      for (int j = 0; j < 8; j++) O[t][j] *= f[j];
    // ---- P (C-layout) -> A-fragment layout via per-wave LDS patch -------
#pragma unroll
    for (int t = 0; t < 4; t++)
#pragma unroll
      for (int j = 0; j < 8; j++)
        myP[(j + half * 8) * 64 + t * 16 + ln16] = f2bf(s[t][j]);
    asm volatile("s_wait_dscnt 0" ::: "memory");
    const bf16_t* pr = myP + ln16 * 64;
    v16bf p0 = load_frag(pr + half * 8,      pr + 16 + half * 8);
    v16bf p1 = load_frag(pr + 32 + half * 8, pr + 48 + half * 8);
    // ---- beta A-fragments (f32 global, float4 loads -> bf16) ------------
    v16bf bb0, bb1;
    {
      const float* bp = Bt + (size_t)(q0 + ln16) * T_ + tk0;
      const float4 c0 = *(const float4*)(bp + half * 8);
      const float4 c1 = *(const float4*)(bp + half * 8 + 4);
      const float4 c2 = *(const float4*)(bp + 16 + half * 8);
      const float4 c3 = *(const float4*)(bp + 16 + half * 8 + 4);
      const float4 c4 = *(const float4*)(bp + 32 + half * 8);
      const float4 c5 = *(const float4*)(bp + 32 + half * 8 + 4);
      const float4 c6 = *(const float4*)(bp + 48 + half * 8);
      const float4 c7 = *(const float4*)(bp + 48 + half * 8 + 4);
      bb0[0]=f2bf(c0.x); bb0[1]=f2bf(c0.y); bb0[2]=f2bf(c0.z); bb0[3]=f2bf(c0.w);
      bb0[4]=f2bf(c1.x); bb0[5]=f2bf(c1.y); bb0[6]=f2bf(c1.z); bb0[7]=f2bf(c1.w);
      bb0[8]=f2bf(c2.x); bb0[9]=f2bf(c2.y); bb0[10]=f2bf(c2.z); bb0[11]=f2bf(c2.w);
      bb0[12]=f2bf(c3.x); bb0[13]=f2bf(c3.y); bb0[14]=f2bf(c3.z); bb0[15]=f2bf(c3.w);
      bb1[0]=f2bf(c4.x); bb1[1]=f2bf(c4.y); bb1[2]=f2bf(c4.z); bb1[3]=f2bf(c4.w);
      bb1[4]=f2bf(c5.x); bb1[5]=f2bf(c5.y); bb1[6]=f2bf(c5.z); bb1[7]=f2bf(c5.w);
      bb1[8]=f2bf(c6.x); bb1[9]=f2bf(c6.y); bb1[10]=f2bf(c6.z); bb1[11]=f2bf(c6.w);
      bb1[12]=f2bf(c7.x); bb1[13]=f2bf(c7.y); bb1[14]=f2bf(c7.z); bb1[15]=f2bf(c7.w);
    }
    // ---- O += P@V ; Bacc += beta@V (V fragments shared) -----------------
#pragma unroll
    for (int dt = 0; dt < 4; dt++) {
      const bf16_t* vr = &sV[(dt * 16 + ln16) * 64];
      v16bf v0 = load_frag(vr + half * 8,      vr + 16 + half * 8);
      v16bf v1 = load_frag(vr + 32 + half * 8, vr + 48 + half * 8);
      O[dt]  = wmma_bf16(p0,  v0, O[dt]);
      O[dt]  = wmma_bf16(p1,  v1, O[dt]);
      Bc[dt] = wmma_bf16(bb0, v0, Bc[dt]);
      Bc[dt] = wmma_bf16(bb1, v1, Bc[dt]);
    }
    __syncthreads();   // all waves done with sK/sV before restaging
  }
  // ---- out = O / l + Bacc ------------------------------------------------
#pragma unroll
  for (int dt = 0; dt < 4; dt++) {
#pragma unroll
    for (int j = 0; j < 8; j++) {
      const int q = q0 + j + half * 8;
      const float v = O[dt][j] / lrun[j] + Bc[dt][j];
      Out[((size_t)b * T_ + q) * DIM_ + h * D_ + dt * 16 + ln16] = v;
    }
  }
}

// ---------------------------------------------------------------------------
extern "C" void kernel_launch(void* const* d_in, const int* in_sizes, int n_in,
                              void* d_out, int out_size, void* d_ws, size_t ws_size,
                              hipStream_t stream) {
  const float* q   = (const float*)d_in[0];
  const float* k   = (const float*)d_in[1];
  const float* v   = (const float*)d_in[2];
  const float* beta = (const float*)d_in[3];
  const unsigned char* src_mask = (const unsigned char*)d_in[4];
  const unsigned char* tgt_mask = (const unsigned char*)d_in[5];
  const float* Wq = (const float*)d_in[6];
  const float* bq = (const float*)d_in[7];
  const float* Wk = (const float*)d_in[8];
  const float* bk = (const float*)d_in[9];
  const float* Wv = (const float*)d_in[10];
  const float* bv = (const float*)d_in[11];
  float* out = (float*)d_out;

  const size_t seg = (size_t)B_ * H_ * T_ * D_;   // 8M bf16 elements (16 MB)
  bf16_t* Qh = (bf16_t*)d_ws;
  bf16_t* Kh = Qh + seg;
  bf16_t* Vt = Kh + seg;

  dim3 pgrid((B_ * T_) / 128, DIM_ / 128);        // 64 x 8
  proj_kernel<<<pgrid, 256, 0, stream>>>(q, Wq, bq, Qh, 0);
  proj_kernel<<<pgrid, 256, 0, stream>>>(k, Wk, bk, Kh, 0);
  proj_kernel<<<pgrid, 256, 0, stream>>>(v, Wv, bv, Vt, 1);

  dim3 agrid(T_ / 128, H_, B_);                   // 8 x 16 x 8
  attn_kernel<<<agrid, 256, 0, stream>>>(Qh, Kh, Vt, beta, src_mask, tgt_mask, out);
}
